// FieldTypedProjector_66949950210382
// MI455X (gfx1250) — matrix operand order repaired and checked by reading, not compiled, not run on hardware
//
#include <hip/hip_runtime.h>
#include <hip/hip_bf16.h>
#include <math.h>

typedef __attribute__((ext_vector_type(16))) _Float16 v16h;
typedef __attribute__((ext_vector_type(8)))  _Float16 v8h;
typedef __attribute__((ext_vector_type(8)))  float    v8f;

#define N_KINDS 8
#define D_MODEL 256
#define B_FEAT  24
#define K1PAD   64      // GEMM1 K padded 48 -> 64
#define TWO_PI  6.28318530717958647692f

// ---------------------------------------------------------------------------
// Pass 1: count tokens per kind
// ---------------------------------------------------------------------------
__global__ __launch_bounds__(256) void ftp_count(const int* __restrict__ kinds,
                                                 int* __restrict__ counts, int T) {
    int t = blockIdx.x * 256 + threadIdx.x;
    if (t < T) atomicAdd(&counts[kinds[t]], 1);
}

// ---------------------------------------------------------------------------
// Pass 2: exclusive scan of 16-padded per-kind counts
// ---------------------------------------------------------------------------
__global__ void ftp_scan(const int* __restrict__ counts, int* __restrict__ offP) {
    if (threadIdx.x == 0) {
        int run = 0;
        for (int k = 0; k < N_KINDS; ++k) {
            offP[k] = run;
            run += ((counts[k] + 15) >> 4) << 4;
        }
        offP[N_KINDS] = run;
    }
}

// ---------------------------------------------------------------------------
// Pass 3: scatter token ids into per-kind segments (padding slots stay -1)
// ---------------------------------------------------------------------------
__global__ __launch_bounds__(256) void ftp_fill(const int* __restrict__ kinds,
                                                const int* __restrict__ offP,
                                                int* __restrict__ cursor,
                                                int* __restrict__ idx, int T) {
    int t = blockIdx.x * 256 + threadIdx.x;
    if (t < T) {
        int k = kinds[t];
        int p = atomicAdd(&cursor[k], 1);
        idx[offP[k] + p] = t;
    }
}

// ---------------------------------------------------------------------------
// Weight packing: f32 row-major [k][K][N]  ->  f16 N-major [k][N][K]
// (K-contiguous so B fragments load as two b128 per lane)
// ---------------------------------------------------------------------------
__global__ __launch_bounds__(256) void ftp_pack_w1(const float* __restrict__ W1,
                                                   _Float16* __restrict__ W1T) {
    int id = blockIdx.x * 256 + threadIdx.x;           // 8*256*64
    int k = id >> 14, rem = id & 16383;
    int K = rem >> 8, n = rem & 255;
    float v = (K < 2 * B_FEAT) ? W1[(k * 2 * B_FEAT + K) * D_MODEL + n] : 0.0f;
    W1T[((size_t)(k * D_MODEL + n)) * K1PAD + K] = (_Float16)v;
}

__global__ __launch_bounds__(256) void ftp_pack_w2(const float* __restrict__ W2,
                                                   _Float16* __restrict__ W2T) {
    int id = blockIdx.x * 256 + threadIdx.x;           // 8*256*256
    int k = id >> 16, rem = id & 65535;
    int K = rem >> 8, n = rem & 255;
    W2T[((size_t)(k * D_MODEL + n)) * D_MODEL + K] =
        (_Float16)W2[((size_t)(k * D_MODEL + K)) * D_MODEL + n];
}

// ---------------------------------------------------------------------------
// Build v16h A/B fragment from two contiguous 8-half vectors
// ---------------------------------------------------------------------------
__device__ __forceinline__ v16h frag16(v8h lo, v8h hi) {
    return __builtin_shufflevector(lo, hi, 0, 1, 2, 3, 4, 5, 6, 7,
                                           8, 9, 10, 11, 12, 13, 14, 15);
}

// ---------------------------------------------------------------------------
// Branchless exact-erf GELU: Abramowitz-Stegun 7.1.26 (|err| <= 1.5e-7,
// far below the f16 GEMM noise floor). v_rcp + v_exp + FMAs, no branches.
// ---------------------------------------------------------------------------
__device__ __forceinline__ float gelu_erf(float x) {
    float z = fabsf(x) * 0.70710678118654752f;          // |x|/sqrt(2)
    float t = __builtin_amdgcn_rcpf(fmaf(0.3275911f, z, 1.0f));
    float p = fmaf(1.061405429f, t, -1.453152027f);
    p = fmaf(p, t, 1.421413741f);
    p = fmaf(p, t, -0.284496736f);
    p = fmaf(p, t, 0.254829592f);
    p = p * t * __expf(-z * z);
    float e = copysignf(1.0f - p, x);                   // erf(x/sqrt(2))
    return 0.5f * x * (1.0f + e);
}

// ---------------------------------------------------------------------------
// Main fused kernel: one block = one 16-token tile of a single kind.
// 256 threads = 8 waves; each wave owns 2 of the 16 N-tiles.
// ---------------------------------------------------------------------------
#define A1P (K1PAD + 8)     // padded LDS row strides (avoid bank conflicts)
#define HP  (D_MODEL + 8)

__global__ __launch_bounds__(256) void ftp_gemm(
    const float* __restrict__ values, const float* __restrict__ Bmat,
    const float* __restrict__ kind_emb,
    const _Float16* __restrict__ W1T, const float* __restrict__ b1v,
    const _Float16* __restrict__ W2T, const float* __restrict__ b2v,
    const int* __restrict__ offP, const int* __restrict__ idx,
    float* __restrict__ out) {

    __shared__ __attribute__((aligned(16))) _Float16 A1[16][A1P];
    __shared__ __attribute__((aligned(16))) _Float16 H[16][HP];
    __shared__ int   sidx[16];
    __shared__ float sval[16];

    const int tid = threadIdx.x;
    const int tileRow = blockIdx.x * 16;

    if (tileRow >= offP[N_KINDS]) return;           // block-uniform exit

    int k = 0;
    #pragma unroll
    for (int kk = 0; kk < N_KINDS - 1; ++kk)
        if (tileRow >= offP[kk + 1]) k = kk + 1;

    if (tid < 16) {
        int t = idx[tileRow + tid];
        sidx[tid] = t;
        sval[tid] = (t >= 0) ? values[t] : 0.0f;
    }
    __syncthreads();

    // Fourier features (48 real + 16 zero pad), 4 slots/thread, hw sin/cos
    {
        int m = tid >> 4, j = tid & 15;
        float v = sval[m];
        #pragma unroll
        for (int q = 0; q < 4; ++q) {
            int f = j * 4 + q;
            float r;
            if (f < B_FEAT)           r = __sinf(TWO_PI * v * Bmat[f]);
            else if (f < 2 * B_FEAT)  r = __cosf(TWO_PI * v * Bmat[f - B_FEAT]);
            else                      r = 0.0f;
            A1[m][f] = (_Float16)r;
        }
    }
    __syncthreads();

    const int wave = tid >> 5;
    const int lane = tid & 31;
    const int lm = lane & 15;   // row (A) / col (B,C,D)
    const int lh = lane >> 4;   // lane-half

    // A fragments for GEMM1: 16-bit A layout is two contiguous 8-half runs
    // (K = lh*8 + 0..7 and K = 16 + lh*8 + 0..7), so read as b128 pairs.
    const v16h a1lo = frag16(*(const v8h*)&A1[lm][lh * 8],
                             *(const v8h*)&A1[lm][16 + lh * 8]);
    const v16h a1hi = frag16(*(const v8h*)&A1[lm][32 + lh * 8],
                             *(const v8h*)&A1[lm][32 + 16 + lh * 8]);

    // ---------------- GEMM1: h = GELU(ff @ W1[k] + b1[k]) ----------------
    #pragma unroll
    for (int nt = 0; nt < 2; ++nt) {
        const int ncol = (wave * 2 + nt) * 16;
        // B layout: lane col = lm, half lh holds K = c*32+lh*8+0..7 and c*32+16+lh*8+0..7
        const _Float16* w1row = W1T + ((size_t)(k * D_MODEL + ncol + lm)) * K1PAD;
        const v16h blo = frag16(*(const v8h*)(w1row + lh * 8),
                                *(const v8h*)(w1row + 16 + lh * 8));
        const v16h bhi = frag16(*(const v8h*)(w1row + 32 + lh * 8),
                                *(const v8h*)(w1row + 32 + 16 + lh * 8));
        v8f c = {};
        c = __builtin_amdgcn_wmma_f32_16x16x32_f16(false, a1lo, false, blo,
                                                   (short)0, c, false, false);
        c = __builtin_amdgcn_wmma_f32_16x16x32_f16(false, a1hi, false, bhi,
                                                   (short)0, c, false, false);
        float bias1 = b1v[k * D_MODEL + ncol + lm];
        #pragma unroll
        for (int r = 0; r < 8; ++r) {
            float g = gelu_erf(c[r] + bias1);
            H[r + 8 * lh][ncol + lm] = (_Float16)g;
        }
    }
    __syncthreads();

    // ---------------- GEMM2: out = h @ W2[k] + b2[k] + kind_emb[k] --------
    #pragma unroll
    for (int nt = 0; nt < 2; ++nt) {
        const int ncol = (wave * 2 + nt) * 16;
        const _Float16* w2row = W2T + ((size_t)(k * D_MODEL + ncol + lm)) * D_MODEL;
        v8f c2 = {};
        #pragma unroll
        for (int kk = 0; kk < 8; ++kk) {
            const v16h a2 = frag16(*(const v8h*)&H[lm][kk * 32 + lh * 8],
                                   *(const v8h*)&H[lm][kk * 32 + 16 + lh * 8]);
            const v16h b2f = frag16(*(const v8h*)(w2row + kk * 32 + lh * 8),
                                    *(const v8h*)(w2row + kk * 32 + 16 + lh * 8));
            c2 = __builtin_amdgcn_wmma_f32_16x16x32_f16(false, a2, false, b2f,
                                                        (short)0, c2, false, false);
        }
        const int col = ncol + lm;
        float bias = b2v[k * D_MODEL + col] + kind_emb[k * D_MODEL + col];
        #pragma unroll
        for (int r = 0; r < 8; ++r) {
            int tok = sidx[r + 8 * lh];
            if (tok >= 0) out[(size_t)tok * D_MODEL + col] = c2[r] + bias;
        }
    }
}

// ---------------------------------------------------------------------------
// ws layout: [0..7] counts | [8..15] cursor | [16..24] offP | idx[T+128]
//            | W1T f16 [8][256][64] | W2T f16 [8][256][256]
// ---------------------------------------------------------------------------
extern "C" void kernel_launch(void* const* d_in, const int* in_sizes, int n_in,
                              void* d_out, int out_size, void* d_ws, size_t ws_size,
                              hipStream_t stream) {
    const float* values   = (const float*)d_in[0];
    const int*   kinds    = (const int*)  d_in[1];
    const float* Bmat     = (const float*)d_in[2];
    const float* kind_emb = (const float*)d_in[3];
    const float* W1       = (const float*)d_in[4];
    const float* b1v      = (const float*)d_in[5];
    const float* W2       = (const float*)d_in[6];
    const float* b2v      = (const float*)d_in[7];
    float* out = (float*)d_out;

    const int T = in_sizes[0];

    int* counts = (int*)d_ws;
    int* cursor = counts + 8;
    int* offP   = counts + 16;
    int* idx    = counts + 32;

    size_t off = 128 + (size_t)(T + 128) * sizeof(int);
    off = (off + 255) & ~(size_t)255;
    _Float16* W1T = (_Float16*)((char*)d_ws + off);
    off += (size_t)N_KINDS * D_MODEL * K1PAD * sizeof(_Float16);
    off = (off + 255) & ~(size_t)255;
    _Float16* W2T = (_Float16*)((char*)d_ws + off);

    hipMemsetAsync(counts, 0, 128, stream);
    hipMemsetAsync(idx, 0xFF, (size_t)(T + 128) * sizeof(int), stream);

    const int nb = (T + 255) / 256;
    ftp_count<<<nb, 256, 0, stream>>>(kinds, counts, T);
    ftp_scan <<<1, 32, 0, stream>>>(counts, offP);
    ftp_fill <<<nb, 256, 0, stream>>>(kinds, offP, cursor, idx, T);

    ftp_pack_w1<<<(N_KINDS * D_MODEL * K1PAD) / 256, 256, 0, stream>>>(W1, W1T);
    ftp_pack_w2<<<(N_KINDS * D_MODEL * D_MODEL) / 256, 256, 0, stream>>>(W2, W2T);

    const int maxTiles = T / 16 + N_KINDS;
    ftp_gemm<<<maxTiles, 256, 0, stream>>>(values, Bmat, kind_emb, W1T, b1v,
                                           W2T, b2v, offP, idx, out);
}
